// MoEExperts_77326591197635
// MI455X (gfx1250) — compile-verified, matrix operand order checked by reading, not run on hardware
//
#include <hip/hip_runtime.h>
#include <math.h>

typedef __bf16 bf16_t;
typedef __attribute__((ext_vector_type(16))) __bf16 v16bf;
typedef __attribute__((ext_vector_type(8)))  __bf16 v8bf;
typedef __attribute__((ext_vector_type(8)))  float  v8f;

// Problem constants (from reference)
constexpr int E_N   = 64;
constexpr int H_N   = 2048;
constexpr int I_N   = 768;
constexpr int T_N   = 4096;
constexpr int K_TOP = 8;
constexpr int CAP_N = 1024;

// Tile config: 256 threads = 8 waves in a 4x2 grid of 32x32 wave-tiles
constexpr int BM = 128;  // rows per block tile
constexpr int BN = 64;   // cols per block tile
constexpr int BK = 32;   // bf16 wmma K

// Workspace layout (bytes)
constexpr size_t WS_COUNTS = 0;                                   // E ints
constexpr size_t WS_TOK    = 1024;                                // E*CAP ints
constexpr size_t WS_WGT    = WS_TOK + (size_t)E_N * CAP_N * 4;    // E*CAP floats
constexpr size_t WS_XB     = WS_WGT + (size_t)E_N * CAP_N * 4;    // T*H bf16
constexpr size_t WS_HBUF   = WS_XB + (size_t)T_N * H_N * 2;       // E*CAP*I bf16

__device__ __forceinline__ v16bf frag16(const bf16_t* p, int hi) {
    // two contiguous 16-byte runs per lane: K = {hi..hi+7} and {hi+16..hi+23}
    v8bf a = *(const v8bf*)(p + hi);
    v8bf b = *(const v8bf*)(p + hi + 16);
    return __builtin_shufflevector(a, b, 0, 1, 2, 3, 4, 5, 6, 7,
                                         8, 9, 10, 11, 12, 13, 14, 15);
}

__device__ __forceinline__ v8bf pack8(const float* f) {
    v8bf r;
    #pragma unroll
    for (int i = 0; i < 8; ++i) r[i] = (bf16_t)f[i];
    return r;
}

// ---------------- dispatch kernels ----------------

__global__ void moe_zero_counts(int* counts) {
    int i = threadIdx.x;
    if (i < E_N) counts[i] = 0;
}

__global__ void moe_assign(const int* __restrict__ sel,
                           const float* __restrict__ rw,
                           int* __restrict__ counts,
                           int* __restrict__ tok,
                           float* __restrict__ wgt) {
    int i = blockIdx.x * blockDim.x + threadIdx.x;
    if (i < T_N * K_TOP) {
        int e = sel[i];
        int slot = atomicAdd(&counts[e], 1);
        if (slot < CAP_N) {
            tok[e * CAP_N + slot] = i / K_TOP;
            wgt[e * CAP_N + slot] = rw[i];
        }
    }
}

__global__ void moe_cvt_x(const float* __restrict__ x, bf16_t* __restrict__ xb) {
    int i = blockIdx.x * blockDim.x + threadIdx.x;
    if (i < T_N * H_N) xb[i] = (bf16_t)x[i];
}

__global__ void moe_zero_out(float* __restrict__ out) {
    int i = blockIdx.x * blockDim.x + threadIdx.x;
    if (i < T_N * H_N) out[i] = 0.0f;
}

// ---------------- GEMM1: x @ [gate|up] -> silu(g)*u -> hbuf ----------------

__global__ __launch_bounds__(256) void moe_gemm1(
    const float* __restrict__ gate, const float* __restrict__ up,
    const bf16_t* __restrict__ xb, const int* __restrict__ counts,
    const int* __restrict__ tok, bf16_t* __restrict__ hbuf)
{
    constexpr int ROWT = CAP_N / BM;   // 8
    constexpr int COLT = I_N / BN;     // 12
    int bid = blockIdx.x;
    int e  = bid / (ROWT * COLT);
    int rt = (bid / COLT) % ROWT;
    int ct = bid % COLT;
    int cnt = counts[e]; if (cnt > CAP_N) cnt = CAP_N;
    int row0 = rt * BM;
    if (row0 >= cnt) return;
    int nt = ct * BN;

    __shared__ __align__(16) bf16_t lA [BM * BK];   // 8 KB
    __shared__ __align__(16) bf16_t lBg[BN * BK];   // 4 KB
    __shared__ __align__(16) bf16_t lBu[BN * BK];   // 4 KB

    int tid  = threadIdx.x;
    int lane = tid & 31;
    int wid  = tid >> 5;
    int wm = wid >> 1, wn = wid & 1;       // 4x2 wave grid -> 32x32 per wave
    int lm = lane & 15;
    int hi = (lane >> 4) * 8;              // K offset / row offset per lane half

    const float* Wg = gate + (size_t)e * H_N * I_N;
    const float* Wu = up   + (size_t)e * H_N * I_N;

    v8f accg[2][2] = {};
    v8f accu[2][2] = {};

    // A-tile assignment: each thread owns one half-row (16 bf16)
    int ar    = tid >> 1;                  // 0..127
    int ahalf = (tid & 1) * 16;
    int arow  = row0 + ar;
    int atok  = (arow < cnt) ? tok[e * CAP_N + arow] : -1;
    const bf16_t* asrc = (atok >= 0) ? (xb + (size_t)atok * H_N + ahalf) : nullptr;
    bf16_t* adst = &lA[ar * BK + ahalf];

    // B-tile assignment: thread owns fixed column n and 8 consecutive K values
    int bn  = tid & 63;
    int bk0 = (tid >> 6) * 8;              // 0,8,16,24
    const float* gsrc = Wg + (size_t)bk0 * I_N + nt + bn;
    const float* usrc = Wu + (size_t)bk0 * I_N + nt + bn;
    bf16_t* gdst = &lBg[bn * BK + bk0];
    bf16_t* udst = &lBu[bn * BK + bk0];

    v8bf aReg[2];
    float gReg[8], uReg[8];

    auto stage = [&](int kk) {
        if (asrc) {
            const v8bf* s = (const v8bf*)(asrc + kk);
            aReg[0] = s[0];
            aReg[1] = s[1];
        } else {
            v8bf z = {};
            aReg[0] = z; aReg[1] = z;
        }
        #pragma unroll
        for (int i = 0; i < 8; ++i) {
            size_t off = (size_t)(kk + i) * I_N;
            gReg[i] = gsrc[off];
            uReg[i] = usrc[off];
        }
    };

    stage(0);
    for (int kk = 0; kk < H_N; kk += BK) {
        // commit staged tile to LDS (A: 2xb128, B: 1xb128 per matrix)
        ((v8bf*)adst)[0] = aReg[0];
        ((v8bf*)adst)[1] = aReg[1];
        *(v8bf*)gdst = pack8(gReg);
        *(v8bf*)udst = pack8(uReg);
        __syncthreads();

        if (kk + BK < H_N) stage(kk + BK);   // overlap next tile's loads

        #pragma unroll
        for (int tm = 0; tm < 2; ++tm) {
            v16bf af = frag16(&lA[(wm * 32 + tm * 16 + lm) * BK], hi);
            #pragma unroll
            for (int tn = 0; tn < 2; ++tn) {
                v16bf bg = frag16(&lBg[(wn * 32 + tn * 16 + lm) * BK], hi);
                v16bf bu = frag16(&lBu[(wn * 32 + tn * 16 + lm) * BK], hi);
                accg[tm][tn] = __builtin_amdgcn_wmma_f32_16x16x32_bf16(
                    false, af, false, bg, (short)0, accg[tm][tn], false, false);
                accu[tm][tn] = __builtin_amdgcn_wmma_f32_16x16x32_bf16(
                    false, af, false, bu, (short)0, accu[tm][tn], false, false);
            }
        }
        __syncthreads();
    }

    // Epilogue: h = silu(g) * u  -> hbuf bf16
    #pragma unroll
    for (int tm = 0; tm < 2; ++tm) {
        #pragma unroll
        for (int tn = 0; tn < 2; ++tn) {
            #pragma unroll
            for (int r = 0; r < 8; ++r) {
                int m = wm * 32 + tm * 16 + r + hi;
                int row = row0 + m;
                if (row < cnt) {
                    int col = nt + wn * 32 + tn * 16 + lm;
                    float g = accg[tm][tn][r];
                    float u = accu[tm][tn][r];
                    float h = g / (1.0f + __expf(-g)) * u;
                    hbuf[(size_t)(e * CAP_N + row) * I_N + col] = (bf16_t)h;
                }
            }
        }
    }
}

// ---------------- GEMM2: hbuf @ down -> scaled scatter-add into out ----------------

__global__ __launch_bounds__(256) void moe_gemm2(
    const float* __restrict__ down, const bf16_t* __restrict__ hbuf,
    const int* __restrict__ counts, const int* __restrict__ tok,
    const float* __restrict__ wgt, float* __restrict__ out)
{
    constexpr int ROWT = CAP_N / BM;   // 8
    constexpr int COLT = H_N / BN;     // 32
    int bid = blockIdx.x;
    int e  = bid / (ROWT * COLT);
    int rt = (bid / COLT) % ROWT;
    int ct = bid % COLT;
    int cnt = counts[e]; if (cnt > CAP_N) cnt = CAP_N;
    int row0 = rt * BM;
    if (row0 >= cnt) return;
    int nt = ct * BN;

    __shared__ __align__(16) bf16_t lA[BM * BK];
    __shared__ __align__(16) bf16_t lB[BN * BK];

    int tid  = threadIdx.x;
    int lane = tid & 31;
    int wid  = tid >> 5;
    int wm = wid >> 1, wn = wid & 1;
    int lm = lane & 15;
    int hi = (lane >> 4) * 8;

    const float* Wd = down + (size_t)e * I_N * H_N;

    v8f acc[2][2] = {};

    int ar    = tid >> 1;
    int ahalf = (tid & 1) * 16;
    int arow  = row0 + ar;
    const bf16_t* asrc = (arow < cnt)
        ? (hbuf + (size_t)(e * CAP_N + arow) * I_N + ahalf) : nullptr;
    bf16_t* adst = &lA[ar * BK + ahalf];

    int bn  = tid & 63;
    int bk0 = (tid >> 6) * 8;
    const float* dsrc = Wd + (size_t)bk0 * H_N + nt + bn;
    bf16_t* ddst = &lB[bn * BK + bk0];

    v8bf aReg[2];
    float dReg[8];

    auto stage = [&](int kk) {
        if (asrc) {
            const v8bf* s = (const v8bf*)(asrc + kk);
            aReg[0] = s[0];
            aReg[1] = s[1];
        } else {
            v8bf z = {};
            aReg[0] = z; aReg[1] = z;
        }
        #pragma unroll
        for (int i = 0; i < 8; ++i)
            dReg[i] = dsrc[(size_t)(kk + i) * H_N];
    };

    stage(0);
    for (int kk = 0; kk < I_N; kk += BK) {
        ((v8bf*)adst)[0] = aReg[0];
        ((v8bf*)adst)[1] = aReg[1];
        *(v8bf*)ddst = pack8(dReg);
        __syncthreads();

        if (kk + BK < I_N) stage(kk + BK);

        #pragma unroll
        for (int tm = 0; tm < 2; ++tm) {
            v16bf af = frag16(&lA[(wm * 32 + tm * 16 + lm) * BK], hi);
            #pragma unroll
            for (int tn = 0; tn < 2; ++tn) {
                v16bf bfr = frag16(&lB[(wn * 32 + tn * 16 + lm) * BK], hi);
                acc[tm][tn] = __builtin_amdgcn_wmma_f32_16x16x32_bf16(
                    false, af, false, bfr, (short)0, acc[tm][tn], false, false);
            }
        }
        __syncthreads();
    }

    // Epilogue: scale by router weight, scatter-add to out[token]
    #pragma unroll
    for (int tm = 0; tm < 2; ++tm) {
        #pragma unroll
        for (int tn = 0; tn < 2; ++tn) {
            #pragma unroll
            for (int r = 0; r < 8; ++r) {
                int m = wm * 32 + tm * 16 + r + hi;
                int row = row0 + m;
                if (row < cnt) {
                    int t = tok[e * CAP_N + row];
                    float w = wgt[e * CAP_N + row];
                    int col = nt + wn * 32 + tn * 16 + lm;
                    atomicAdd(&out[(size_t)t * H_N + col], w * acc[tm][tn][r]);
                }
            }
        }
    }
}

// ---------------- launcher ----------------

extern "C" void kernel_launch(void* const* d_in, const int* in_sizes, int n_in,
                              void* d_out, int out_size, void* d_ws, size_t ws_size,
                              hipStream_t stream) {
    const float* x    = (const float*)d_in[0];   // [T,H]
    const float* rw   = (const float*)d_in[1];   // [T,K]
    const int*   sel  = (const int*)d_in[2];     // [T,K]
    const float* gate = (const float*)d_in[3];   // [E,H,I]
    const float* up   = (const float*)d_in[4];   // [E,H,I]
    const float* down = (const float*)d_in[5];   // [E,I,H]
    float* out = (float*)d_out;                  // [T,H]

    char* ws = (char*)d_ws;
    int*    counts = (int*)(ws + WS_COUNTS);
    int*    tok    = (int*)(ws + WS_TOK);
    float*  wgt    = (float*)(ws + WS_WGT);
    bf16_t* xb     = (bf16_t*)(ws + WS_XB);
    bf16_t* hbuf   = (bf16_t*)(ws + WS_HBUF);

    moe_zero_counts<<<1, 64, 0, stream>>>(counts);
    moe_assign<<<(T_N * K_TOP + 255) / 256, 256, 0, stream>>>(sel, rw, counts, tok, wgt);
    moe_cvt_x<<<(T_N * H_N + 255) / 256, 256, 0, stream>>>(x, xb);
    moe_zero_out<<<(T_N * H_N + 255) / 256, 256, 0, stream>>>(out);

    int g1 = E_N * (CAP_N / BM) * (I_N / BN);   // 64*8*12 = 6144
    moe_gemm1<<<g1, 256, 0, stream>>>(gate, up, xb, counts, tok, hbuf);

    int g2 = E_N * (CAP_N / BM) * (H_N / BN);   // 64*8*32 = 16384
    moe_gemm2<<<g2, 256, 0, stream>>>(down, hbuf, counts, tok, wgt, out);
}
